// TransformerDecoder_tcn_86723979641368
// MI455X (gfx1250) — matrix operand order repaired and checked by reading, not compiled, not run on hardware
//
#include <hip/hip_runtime.h>
#include <hip/hip_bf16.h>

// Graph TransformerConv x3 + FC for MI455X (gfx1250, wave32).
// GEMMs: f16-preconverted A (row-major, K padded to 32) + weight pre-swizzled
// into WMMA B-fragment layout -> inner loop is 3 vector loads + 1 wmma.
// Edge attention: L2-resident float4 gathers + u32-key atomicMax / f32
// atomicAdd segment softmax.

#define NN 50000
#define EE 800000
#define HH 4

typedef __attribute__((ext_vector_type(16))) _Float16 v16h;
typedef __attribute__((ext_vector_type(8)))  _Float16 v8h;
typedef __attribute__((ext_vector_type(8)))  float    v8f;

__device__ __forceinline__ unsigned fkey(float f) {
  unsigned b = __float_as_uint(f);
  return (b & 0x80000000u) ? ~b : (b | 0x80000000u);
}
__device__ __forceinline__ float funkey(unsigned k) {
  return __uint_as_float((k & 0x80000000u) ? (k ^ 0x80000000u) : ~k);
}

// ---- activations f32 -> f16, row-padded from K to Kpad (Kpad % 32 == 0) ----
__global__ void cvt_pad_f16(const float* __restrict__ in, _Float16* __restrict__ out,
                            int K, int Kpad)
{
  const int row = blockIdx.y;
  const int c = threadIdx.x + blockIdx.x * blockDim.x;
  if (c < Kpad)
    out[(size_t)row * Kpad + c] =
        (c < K) ? (_Float16)in[(size_t)row * K + c] : (_Float16)0.f;
}

// ---- weights K x Nc f32 -> f16 in B-fragment layout -------------------------
// Block (nTile, ks): 32 lanes x 16 halves contiguous.
// lane L: col = nTile*16 + (L&15), K = ks*32 + (L>>4)*16 + e
__global__ void prep_w_h(const float* __restrict__ W, _Float16* __restrict__ Wf,
                         int K, int Kpad, int Nc)
{
  const int kSteps = Kpad >> 5;
  const int total = (Nc >> 4) * kSteps * 512;
  for (int t = blockIdx.x * blockDim.x + threadIdx.x; t < total;
       t += gridDim.x * blockDim.x) {
    const int e    = t & 15;
    const int lane = (t >> 4) & 31;
    const int tile = t >> 9;
    const int ks   = tile % kSteps;
    const int nt   = tile / kSteps;
    const int col  = (nt << 4) + (lane & 15);
    const int kk   = (ks << 5) + ((lane >> 4) << 4) + e;
    Wf[t] = (kk < K) ? (_Float16)W[(size_t)kk * Nc + col] : (_Float16)0.f;
  }
}

// ---- WMMA GEMM: Out[M,Nc] = act(A @ W + bias). M % 16 == 0, Nc % 16 == 0. --
// act: 0 = none, 1 = leaky_relu(0.01), 2 = relu. One wave -> one 16x16 tile.
__global__ void gemm_wmma_h(const _Float16* __restrict__ A,
                            const _Float16* __restrict__ Wf,
                            const float* __restrict__ bias, float* __restrict__ Out,
                            int M, int Kpad, int Nc, int act)
{
  const int lane  = threadIdx.x;                        // 0..31
  const int mTile = blockIdx.x * blockDim.y + threadIdx.y;
  const int nTile = blockIdx.y;
  const int m0 = mTile << 4;
  if (m0 >= M) return;                                  // wave-uniform
  const int half = lane >> 4, l16 = lane & 15;
  const int kSteps = Kpad >> 5;
  const _Float16* arow = A + (size_t)(m0 + l16) * Kpad + half * 16;
  const _Float16* bptr = Wf + ((size_t)(nTile * kSteps) * 32 + lane) * 16;

  v8f acc = {};
  for (int ks = 0; ks < kSteps; ++ks) {
    v8h alo = *(const v8h*)(arow);            // K = k0 + half*16 + 0..7
    v8h ahi = *(const v8h*)(arow + 8);        // K = k0 + half*16 + 8..15
    v16h a = __builtin_shufflevector(alo, ahi,
        0,1,2,3,4,5,6,7,8,9,10,11,12,13,14,15);
    v16h b = *(const v16h*)(bptr);            // contiguous, coalesced
    acc = __builtin_amdgcn_wmma_f32_16x16x32_f16(false, a, false, b,
                                                 (short)0, acc, false, false);
    arow += 32;
    bptr += 512;
  }
  const int col = (nTile << 4) + l16;
  const float bv = bias ? bias[col] : 0.f;
#pragma unroll
  for (int r = 0; r < 8; ++r) {               // row = m0 + r + 8*half
    float vv = acc[r] + bv;
    if (act == 1)      vv = (vv > 0.f) ? vv : 0.01f * vv;
    else if (act == 2) vv = fmaxf(vv, 0.f);
    Out[(size_t)(m0 + r + (half << 3)) * Nc + col] = vv;
  }
}

// ------------- Edge pass 1: logits + per-(dst,head) running max -------------
__global__ void edge_logits_max(const int* __restrict__ src, const int* __restrict__ dst,
                                const float* __restrict__ ea, const float* __restrict__ q,
                                const float* __restrict__ k,  const float* __restrict__ We,
                                float* __restrict__ logits, unsigned* __restrict__ nmax,
                                int C)
{
  extern __shared__ float sWe[];                 // [HH][HC]
  const int HC = HH * C;
  for (int i = threadIdx.x; i < HH * HC; i += blockDim.x) sWe[i] = We[i];
  __syncthreads();
  const float inv = rsqrtf((float)C);
  for (int e = blockIdx.x * blockDim.x + threadIdx.x; e < EE;
       e += gridDim.x * blockDim.x) {
    const int s = src[e], d = dst[e];
    const float a0 = ea[(size_t)e*HH+0], a1 = ea[(size_t)e*HH+1];
    const float a2 = ea[(size_t)e*HH+2], a3 = ea[(size_t)e*HH+3];
    const float* krow = k + (size_t)s * HC;
    const float* qrow = q + (size_t)d * HC;
#pragma unroll
    for (int h = 0; h < HH; ++h) {
      const float4* qp = (const float4*)(qrow + h * C);
      const float4* kp = (const float4*)(krow + h * C);
      const float4* w0 = (const float4*)(sWe + h * C);
      const float4* w1 = (const float4*)(sWe + HC + h * C);
      const float4* w2 = (const float4*)(sWe + 2 * HC + h * C);
      const float4* w3 = (const float4*)(sWe + 3 * HC + h * C);
      float acc = 0.f;
      for (int c4 = 0; c4 < (C >> 2); ++c4) {
        float4 qv = qp[c4], kv = kp[c4];
        float4 e0 = w0[c4], e1 = w1[c4], e2 = w2[c4], e3 = w3[c4];
        acc += qv.x * (kv.x + a0*e0.x + a1*e1.x + a2*e2.x + a3*e3.x);
        acc += qv.y * (kv.y + a0*e0.y + a1*e1.y + a2*e2.y + a3*e3.y);
        acc += qv.z * (kv.z + a0*e0.z + a1*e1.z + a2*e2.z + a3*e3.z);
        acc += qv.w * (kv.w + a0*e0.w + a1*e1.w + a2*e2.w + a3*e3.w);
      }
      float lg = acc * inv;
      logits[(size_t)e * HH + h] = lg;
      atomicMax(&nmax[(size_t)d * HH + h], fkey(lg));
    }
  }
}

// ------------- Edge pass 2: softmax denominator -----------------------------
__global__ void edge_den(const int* __restrict__ dst, const float* __restrict__ logits,
                         const unsigned* __restrict__ nmax, float* __restrict__ den)
{
  for (int e = blockIdx.x * blockDim.x + threadIdx.x; e < EE;
       e += gridDim.x * blockDim.x) {
    const int d = dst[e];
#pragma unroll
    for (int h = 0; h < HH; ++h) {
      float ex = expf(logits[(size_t)e*HH + h] - funkey(nmax[(size_t)d*HH + h]));
      atomicAdd(&den[(size_t)d * HH + h], ex);
    }
  }
}

// ------------- Edge pass 3: alpha + scatter-add messages --------------------
__global__ void edge_message(const int* __restrict__ src, const int* __restrict__ dst,
                             const float* __restrict__ ea, const float* __restrict__ v,
                             const float* __restrict__ We, const float* __restrict__ logits,
                             const unsigned* __restrict__ nmax, const float* __restrict__ den,
                             float* __restrict__ alphaOut, float* __restrict__ agg, int C)
{
  extern __shared__ float sWe[];
  const int HC = HH * C;
  for (int i = threadIdx.x; i < HH * HC; i += blockDim.x) sWe[i] = We[i];
  __syncthreads();
  for (int e = blockIdx.x * blockDim.x + threadIdx.x; e < EE;
       e += gridDim.x * blockDim.x) {
    const int s = src[e], d = dst[e];
    const float a0 = ea[(size_t)e*HH+0], a1 = ea[(size_t)e*HH+1];
    const float a2 = ea[(size_t)e*HH+2], a3 = ea[(size_t)e*HH+3];
    const float* vrow = v + (size_t)s * HC;
#pragma unroll
    for (int h = 0; h < HH; ++h) {
      const float m  = funkey(nmax[(size_t)d*HH + h]);
      const float dn = den[(size_t)d*HH + h];
      const float alpha = expf(logits[(size_t)e*HH + h] - m) / (dn + 1e-16f);
      alphaOut[(size_t)e * HH + h] = alpha;
      const float4* vp = (const float4*)(vrow + h * C);
      const float4* w0 = (const float4*)(sWe + h * C);
      const float4* w1 = (const float4*)(sWe + HC + h * C);
      const float4* w2 = (const float4*)(sWe + 2 * HC + h * C);
      const float4* w3 = (const float4*)(sWe + 3 * HC + h * C);
      float* ap = agg + (size_t)d * HC + h * C;
      for (int c4 = 0; c4 < (C >> 2); ++c4) {
        float4 vv = vp[c4];
        float4 e0 = w0[c4], e1 = w1[c4], e2 = w2[c4], e3 = w3[c4];
        atomicAdd(ap + 4*c4 + 0, (vv.x + a0*e0.x + a1*e1.x + a2*e2.x + a3*e3.x) * alpha);
        atomicAdd(ap + 4*c4 + 1, (vv.y + a0*e0.y + a1*e1.y + a2*e2.y + a3*e3.y) * alpha);
        atomicAdd(ap + 4*c4 + 2, (vv.z + a0*e0.z + a1*e1.z + a2*e2.z + a3*e3.z) * alpha);
        atomicAdd(ap + 4*c4 + 3, (vv.w + a0*e0.w + a1*e1.w + a2*e2.w + a3*e3.w) * alpha);
      }
    }
  }
}

// ------------- Beta-gated skip: out2 = beta*xr + (1-beta)*agg ---------------
__global__ void node_gate(const float* __restrict__ agg, const float* __restrict__ xr,
                          const float* __restrict__ Wb, float* __restrict__ out2, int C)
{
  extern __shared__ float sWb[];                 // [3*HC]
  const int HC = HH * C;
  for (int i = threadIdx.x; i < 3 * HC; i += blockDim.x) sWb[i] = Wb[i];
  __syncthreads();
  for (int n = blockIdx.x * blockDim.x + threadIdx.x; n < NN;
       n += gridDim.x * blockDim.x) {
    const float4* o4 = (const float4*)(agg + (size_t)n * HC);
    const float4* x4 = (const float4*)(xr  + (size_t)n * HC);
    const float4* b0 = (const float4*)(sWb);
    const float4* b1 = (const float4*)(sWb + HC);
    const float4* b2 = (const float4*)(sWb + 2 * HC);
    float s = 0.f;
    for (int c4 = 0; c4 < (HC >> 2); ++c4) {
      float4 o = o4[c4], x = x4[c4];
      float4 w0 = b0[c4], w1 = b1[c4], w2 = b2[c4];
      s += o.x*w0.x + x.x*w1.x + (o.x - x.x)*w2.x;
      s += o.y*w0.y + x.y*w1.y + (o.y - x.y)*w2.y;
      s += o.z*w0.z + x.z*w1.z + (o.z - x.z)*w2.z;
      s += o.w*w0.w + x.w*w1.w + (o.w - x.w)*w2.w;
    }
    const float beta = 1.f / (1.f + expf(-s));
    float4* w4 = (float4*)(out2 + (size_t)n * HC);
    for (int c4 = 0; c4 < (HC >> 2); ++c4) {
      float4 o = o4[c4], x = x4[c4], r;
      r.x = beta*x.x + (1.f-beta)*o.x;  r.y = beta*x.y + (1.f-beta)*o.y;
      r.z = beta*x.z + (1.f-beta)*o.z;  r.w = beta*x.w + (1.f-beta)*o.w;
      w4[c4] = r;
    }
  }
}

__global__ void cast_i32_f32(const int* __restrict__ in, float* __restrict__ out, int n)
{
  for (int i = blockIdx.x * blockDim.x + threadIdx.x; i < n;
       i += gridDim.x * blockDim.x)
    out[i] = (float)in[i];
}

extern "C" void kernel_launch(void* const* d_in, const int* in_sizes, int n_in,
                              void* d_out, int out_size, void* d_ws, size_t ws_size,
                              hipStream_t stream)
{
  (void)in_sizes; (void)n_in; (void)out_size; (void)ws_size;
  const float* z0  = (const float*)d_in[0];
  const int*   ei  = (const int*)d_in[1];
  const float* ea0 = (const float*)d_in[2];
  const int* srcp = ei;            // edge_index[0]
  const int* dstp = ei + EE;       // edge_index[1]

  // ---- workspace carve-out. Max HC = 192. ----
  float* ws = (float*)d_ws;
  size_t o = 0;
  const size_t HCmax = (size_t)NN * 192;
  float* q      = ws + o; o += HCmax;
  float* kbuf   = ws + o; o += HCmax;
  float* v      = ws + o; o += HCmax;
  float* xr     = ws + o; o += HCmax;
  float* zA     = ws + o; o += (size_t)NN * 48;
  float* zB     = ws + o; o += (size_t)NN * 48;
  float* logits = ws + o; o += (size_t)EE * HH;
  float* alpha1 = ws + o; o += (size_t)EE * HH;
  float* alpha2 = ws + o; o += (size_t)EE * HH;
  float* den    = ws + o; o += (size_t)NN * HH;
  unsigned* nmax = (unsigned*)(ws + o); o += (size_t)NN * HH;
  _Float16* Ah  = (_Float16*)(ws + o); o += HCmax / 2 + 64;  // N x 192 halfs
  _Float16* Wf[4];
  for (int i = 0; i < 4; ++i) { Wf[i] = (_Float16*)(ws + o); o += 24576 + 64; }
  float* agg  = kbuf;  // K dead after edge_logits_max; zeroed after it runs
  float* out2 = q;     // Q dead after edge_logits_max

  float* outZ     = (float*)d_out;                    // [NN,16]
  float* outEdges = outZ + (size_t)NN * 16;           // [2,EE] as float
  float* outAlpha = outEdges + 2 * (size_t)EE;        // [EE,HH]

  const int dins[3] = {64, 16, 32};
  const int Cs[3]   = {16, 32, 48};
  const dim3 gb(32, 4);                               // 4 waves / block
  const int mBlocks = (NN / 16 + 3) / 4;              // 3125 M-tiles

  const float* zin = z0;
  const float* eaIn = ea0;
  float* zNexts[3] = {zA, zB, zA};

  auto pad32 = [](int k) { return (k + 31) & ~31; };

  for (int l = 0; l < 3; ++l) {
    const int din = dins[l], C = Cs[l], HC = HH * C;
    const int dinP = pad32(din);
    const int b = 3 + 12 * l;                         // params in layer dict order
    const float* Wq = (const float*)d_in[b + 0];
    const float* bq = (const float*)d_in[b + 1];
    const float* Wk = (const float*)d_in[b + 2];
    const float* bk = (const float*)d_in[b + 3];
    const float* Wv = (const float*)d_in[b + 4];
    const float* bv = (const float*)d_in[b + 5];
    const float* We = (const float*)d_in[b + 6];
    const float* Wsk = (const float*)d_in[b + 7];
    const float* bsk = (const float*)d_in[b + 8];
    const float* Wbeta = (const float*)d_in[b + 9];
    const float* Wt = (const float*)d_in[b + 10];
    const float* bt = (const float*)d_in[b + 11];

    // A (activations) -> f16 padded; weights -> fragment layout
    cvt_pad_f16<<<dim3(1, NN), 256, 0, stream>>>(zin, Ah, din, dinP);
    const int wElems = (HC / 16) * (dinP / 32) * 512;
    prep_w_h<<<(wElems + 255) / 256, 256, 0, stream>>>(Wq,  Wf[0], din, dinP, HC);
    prep_w_h<<<(wElems + 255) / 256, 256, 0, stream>>>(Wk,  Wf[1], din, dinP, HC);
    prep_w_h<<<(wElems + 255) / 256, 256, 0, stream>>>(Wv,  Wf[2], din, dinP, HC);
    prep_w_h<<<(wElems + 255) / 256, 256, 0, stream>>>(Wsk, Wf[3], din, dinP, HC);

    dim3 gg(mBlocks, HC / 16);
    gemm_wmma_h<<<gg, gb, 0, stream>>>(Ah, Wf[0], bq,  q,    NN, dinP, HC, 0);
    gemm_wmma_h<<<gg, gb, 0, stream>>>(Ah, Wf[1], bk,  kbuf, NN, dinP, HC, 0);
    gemm_wmma_h<<<gg, gb, 0, stream>>>(Ah, Wf[2], bv,  v,    NN, dinP, HC, 0);
    gemm_wmma_h<<<gg, gb, 0, stream>>>(Ah, Wf[3], bsk, xr,   NN, dinP, HC, 0);

    hipMemsetAsync(nmax, 0, (size_t)NN * HH * sizeof(unsigned), stream);
    hipMemsetAsync(den,  0, (size_t)NN * HH * sizeof(float), stream);

    const int ldsWe = HH * HC * (int)sizeof(float);
    edge_logits_max<<<EE / 256, 256, ldsWe, stream>>>(srcp, dstp, eaIn, q, kbuf,
                                                      We, logits, nmax, C);
    hipMemsetAsync(agg, 0, (size_t)NN * HC * sizeof(float), stream); // after E1 (alias of K)
    edge_den<<<EE / 256, 256, 0, stream>>>(dstp, logits, nmax, den);

    float* aOut = (l == 2) ? outAlpha : (l == 0 ? alpha1 : alpha2);
    edge_message<<<EE / 256, 256, ldsWe, stream>>>(srcp, dstp, eaIn, v, We,
                                                   logits, nmax, den, aOut, agg, C);

    node_gate<<<(NN + 255) / 256, 256, 3 * HC * (int)sizeof(float), stream>>>(
        agg, xr, Wbeta, out2, C);

    // out2 @ Wt + bt, leaky_relu  (HC already % 32 == 0)
    cvt_pad_f16<<<dim3(1, NN), 256, 0, stream>>>(out2, Ah, HC, HC);
    const int wtElems = (C / 16) * (HC / 32) * 512;
    prep_w_h<<<(wtElems + 255) / 256, 256, 0, stream>>>(Wt, Wf[0], HC, HC, C);
    dim3 gt(mBlocks, C / 16);
    gemm_wmma_h<<<gt, gb, 0, stream>>>(Ah, Wf[0], bt, zNexts[l], NN, HC, C, 1);

    zin = zNexts[l];
    eaIn = aOut;
  }

  // Final FC + ReLU straight into d_out, then edge_index as float
  const float* Wfc = (const float*)d_in[3 + 12 * 3];
  const float* bfc = (const float*)d_in[3 + 12 * 3 + 1];
  cvt_pad_f16<<<dim3(1, NN), 256, 0, stream>>>(zin, Ah, 48, 64);
  prep_w_h<<<(1 * 2 * 512 + 255) / 256, 256, 0, stream>>>(Wfc, Wf[0], 48, 64, 16);
  dim3 gf(mBlocks, 1);
  gemm_wmma_h<<<gf, gb, 0, stream>>>(Ah, Wf[0], bfc, outZ, NN, 64, 16, 2);
  cast_i32_f32<<<(2 * EE + 255) / 256, 256, 0, stream>>>(ei, outEdges, 2 * EE);
}
